// Decoder_78915729096963
// MI455X (gfx1250) — matrix operand order repaired and checked by reading, not compiled, not run on hardware
//
#include <hip/hip_runtime.h>
#include <hip/hip_bf16.h>

// ---------------------------------------------------------------------------
// B=32, S=64, L=512, H=768, D3=128, E_DIM=768.
//   hdc   : [16384,128] = context @ Wh + bh            (parallel, once)
//   h_all : [65,32,768]  serial recurrence: ONE workgroup, h resident in LDS
//   q     : [2048,128]  = h_all[1:] @ Wi + bi          (parallel, once)
//   atts  : [64,32,1,512] = sum_d V[d]*tanh(q+hdc)     (parallel, once)
//   prts  : [64,32] = argmax(softmax over size-1 axis -> ones) == 0
// ---------------------------------------------------------------------------

typedef __attribute__((ext_vector_type(16))) _Float16 v16h;
typedef __attribute__((ext_vector_type(8)))  _Float16 v8h;
typedef __attribute__((ext_vector_type(8)))  float    v8f;

// Branch-free tanh: 1 - 2/(exp(2x)+1). exp->inf gives +1, exp->0 gives -1.
// Straight-line v_exp_f32 + v_rcp_f32; no exec-mask divergence.
__device__ __forceinline__ float tanh_fast(float x)
{
  const float e = __expf(2.0f * x);
  return 1.0f - __fdividef(2.0f, e + 1.0f);
}

// WMMA fragment layouts (ISA 7.12.2, wave32):
//  A (16x32 f16): lane l holds row m=l&15; element e -> K = (e<8?e:e+8)+8*(l>=16)
//    => two contiguous 8-element runs: [lhi*8, +8) and [16+lhi*8, +8).
//  B (32x16 f16): element e -> K = (e<8?e:e+8)+8*(l>=16), col n = l&15.
//  C/D (16x16 f32): VGPR r -> row m = r + 8*(l>=16), col n = l&15.

// ---------------------------------------------------------------------------
// Pack B [K,N] f32 row-major -> f16 B-fragment order (contiguous 32B / lane).
// ---------------------------------------------------------------------------
__global__ void pack_b_f16(const float* __restrict__ Bm,
                           _Float16* __restrict__ Bp, int K, int N)
{
  const int p = blockIdx.x * blockDim.x + threadIdx.x;
  if (p >= K * N) return;
  const int e    = p & 15;
  const int lane = (p >> 4) & 31;
  const int rest = p >> 9;
  const int tilesN = N >> 4;
  const int ct = rest % tilesN;
  const int kb = rest / tilesN;
  const int k = kb * 32 + (e & 7) + ((lane >> 4) << 3) + ((e >> 3) << 4);
  const int n = ct * 16 + (lane & 15);
  Bp[p] = (_Float16)Bm[(size_t)k * N + n];
}

// A fragment from f32 memory (two contiguous 8-float runs -> 4x b128 + cvt).
__device__ __forceinline__ v16h load_a_frag_f32(const float* __restrict__ src,
                                                int lhi)
{
  const float4 r0 = *(const float4*)(src + (lhi << 3));
  const float4 r1 = *(const float4*)(src + (lhi << 3) + 4);
  const float4 r2 = *(const float4*)(src + 16 + (lhi << 3));
  const float4 r3 = *(const float4*)(src + 16 + (lhi << 3) + 4);
  v16h a;
  a[0]=(_Float16)r0.x;  a[1]=(_Float16)r0.y;  a[2]=(_Float16)r0.z;  a[3]=(_Float16)r0.w;
  a[4]=(_Float16)r1.x;  a[5]=(_Float16)r1.y;  a[6]=(_Float16)r1.z;  a[7]=(_Float16)r1.w;
  a[8]=(_Float16)r2.x;  a[9]=(_Float16)r2.y;  a[10]=(_Float16)r2.z; a[11]=(_Float16)r2.w;
  a[12]=(_Float16)r3.x; a[13]=(_Float16)r3.y; a[14]=(_Float16)r3.z; a[15]=(_Float16)r3.w;
  return a;
}

// A fragment from f16 LDS (two 16B runs -> 2x ds_load_b128, no conversion).
__device__ __forceinline__ v16h load_a_frag_lds(const _Float16* src, int lhi)
{
  const v8h lo = *(const v8h*)(src + (lhi << 3));
  const v8h hi = *(const v8h*)(src + 16 + (lhi << 3));
  v16h a;
#pragma unroll
  for (int e = 0; e < 8; ++e) { a[e] = lo[e]; a[e + 8] = hi[e]; }
  return a;
}

// ---------------------------------------------------------------------------
// WMMA GEMM with packed-f16 B: C[M,N] = A[M,K] * B + bias.
// One wave computes a 16 x (16*CT) strip; A fragment reused across CT wmma.
// ---------------------------------------------------------------------------
template<int CT>
__global__ void wmma_gemm_pk(const float* __restrict__ A, int lda,
                             const _Float16* __restrict__ Bp,
                             const float* __restrict__ bias,
                             float* __restrict__ C,
                             int M, int N, int K)
{
  const int wave = threadIdx.x >> 5;
  const int lane = threadIdx.x & 31;
  const int lhi  = lane >> 4;
  const int tilesN    = N >> 4;
  const int colGroups = tilesN / CT;
  const int numWaves  = (M >> 4) * colGroups;

  const int w = blockIdx.x * (blockDim.x >> 5) + wave;
  if (w >= numWaves) return;                  // wave-uniform

  const int row0   = (w / colGroups) << 4;
  const int ctBase = (w % colGroups) * CT;
  const int arow   = row0 + (lane & 15);
  const int ncol   = lane & 15;

  v8f acc[CT];
#pragma unroll
  for (int j = 0; j < CT; ++j) acc[j] = v8f{0.f,0.f,0.f,0.f,0.f,0.f,0.f,0.f};

  const int KB = K >> 5;
  for (int kb = 0; kb < KB; ++kb) {
    const float* aSrc = A + (size_t)arow * lda + (kb << 5);
    if (kb + 1 < KB) __builtin_prefetch(aSrc + 32, 0, 1);
    const v16h a = load_a_frag_f32(aSrc, lhi);

    const _Float16* bBase =
        Bp + (((size_t)kb * tilesN + ctBase) * 32 + lane) * 16;
#pragma unroll
    for (int j = 0; j < CT; ++j) {
      const v16h b = *(const v16h*)(bBase + (size_t)j * 32 * 16);
      acc[j] = __builtin_amdgcn_wmma_f32_16x16x32_f16(false, a, false, b,
                                                      (short)0, acc[j],
                                                      false, false);
    }
  }

#pragma unroll
  for (int j = 0; j < CT; ++j) {
    const int n  = (ctBase + j) * 16 + ncol;
    const float bn = bias[n];
#pragma unroll
    for (int r = 0; r < 8; ++r) {
      const int m = row0 + r + (lhi << 3);
      C[(size_t)m * N + n] = acc[j][r] + bn;
    }
  }
}

// ---------------------------------------------------------------------------
// Entire 65-step recurrence in ONE workgroup (24 waves, 768 threads).
// h kept in LDS as f16 (it is consumed as f16 WMMA A-fragments anyway);
// each step: K<768 half of A from global (emb gather / first_emb),
//            K>=768 half from LDS via ds_load_b128.
// h_all (f32) is streamed to global for the later q-GEMM.
// LDS row pitch 776 halfs keeps b128 LDS reads spread across banks.
// ---------------------------------------------------------------------------
#define HPITCH 776

__global__ void
__launch_bounds__(768)
wmma_step_all(const int* __restrict__ ids,
              const float* __restrict__ emb,
              const float* __restrict__ femb,
              const float* __restrict__ fhid,
              const _Float16* __restrict__ WdP,
              const float* __restrict__ bd,
              float* __restrict__ h_all)
{
  __shared__ _Float16 hbuf[32 * HPITCH];      // 48.5 KB

  constexpr int CT = 4;
  const int N = 768, tilesN = 48, colGroups = 12;
  const int w    = threadIdx.x >> 5;          // 0..23
  const int lane = threadIdx.x & 31;
  const int lhi  = lane >> 4;

  const int row0   = (w / colGroups) << 4;    // 0 or 16
  const int ctBase = (w % colGroups) * CT;
  const int arow   = row0 + (lane & 15);      // batch index b of A rows
  const int ncol   = lane & 15;

  // Preload first_hidden -> LDS (f16).
  for (int i = threadIdx.x; i < 32 * 768; i += 768) {
    const int m = i / 768, k = i - m * 768;
    hbuf[m * HPITCH + k] = (_Float16)fhid[i];
  }
  __syncthreads();

  for (int st = -1; st < 64; ++st) {
    const float* inpRow =
        (st < 0) ? femb : (emb + (size_t)ids[arow * 64 + st] * 768);

    // Prefetch next step's embedding row while this step computes.
    if (st + 1 < 64) {
      const float* nxt = emb + (size_t)ids[arow * 64 + (st + 1)] * 768;
      __builtin_prefetch(nxt + ((lane & 15) << 5) + ((lhi) << 4), 0, 1);
    }

    v8f acc[CT];
#pragma unroll
    for (int j = 0; j < CT; ++j) acc[j] = v8f{0.f,0.f,0.f,0.f,0.f,0.f,0.f,0.f};

    // K = 0..767 : input/embedding half (global, f32 -> f16)
    for (int kb = 0; kb < 24; ++kb) {
      const v16h a = load_a_frag_f32(inpRow + (kb << 5), lhi);
      const _Float16* bBase =
          WdP + (((size_t)kb * tilesN + ctBase) * 32 + lane) * 16;
#pragma unroll
      for (int j = 0; j < CT; ++j) {
        const v16h b = *(const v16h*)(bBase + (size_t)j * 32 * 16);
        acc[j] = __builtin_amdgcn_wmma_f32_16x16x32_f16(false, a, false, b,
                                                        (short)0, acc[j],
                                                        false, false);
      }
    }
    // K = 768..1535 : previous hidden state (LDS f16, ds_load_b128)
    for (int kb = 24; kb < 48; ++kb) {
      const v16h a =
          load_a_frag_lds(&hbuf[arow * HPITCH + ((kb - 24) << 5)], lhi);
      const _Float16* bBase =
          WdP + (((size_t)kb * tilesN + ctBase) * 32 + lane) * 16;
#pragma unroll
      for (int j = 0; j < CT; ++j) {
        const v16h b = *(const v16h*)(bBase + (size_t)j * 32 * 16);
        acc[j] = __builtin_amdgcn_wmma_f32_16x16x32_f16(false, a, false, b,
                                                        (short)0, acc[j],
                                                        false, false);
      }
    }

    __syncthreads();   // all reads of hbuf done before overwrite

    float* hOut = h_all + (size_t)(st + 1) * 32 * 768;
#pragma unroll
    for (int j = 0; j < CT; ++j) {
      const int n  = (ctBase + j) * 16 + ncol;
      const float bn = bd[n];
#pragma unroll
      for (int r = 0; r < 8; ++r) {
        const int m = row0 + r + (lhi << 3);
        const float v = tanh_fast(acc[j][r] + bn);
        hbuf[m * HPITCH + n]       = (_Float16)v;
        hOut[(size_t)m * 768 + n]  = v;
      }
    }
    __syncthreads();   // new h visible before next step's reads
  }
}

// ---------------------------------------------------------------------------
// Fused attention: one wave per (b,l); hdc row + V pinned in registers,
// loop over all 64 steps streaming the (shared, cache-hot) q rows.
//   atts[s,b,l] = sum_d V[d] * tanh(q[s*32+b,d] + hdc[b*512+l,d])
// ---------------------------------------------------------------------------
__global__ void attn_kernel(const float* __restrict__ q,
                            const float* __restrict__ hdc,
                            const float* __restrict__ V,
                            float* __restrict__ atts)
{
  const int wave = threadIdx.x >> 5;
  const int lane = threadIdx.x & 31;
  const unsigned w = blockIdx.x * (blockDim.x >> 5) + wave;  // 0..16383 (b,l)
  if (w >= 32u * 512u) return;

  const int b  = w >> 9;
  const int d0 = lane << 2;

  const float4 hv = *(const float4*)(hdc + ((size_t)w << 7) + d0);
  const float4 vv = *(const float4*)(V + d0);

  for (int s = 0; s < 64; ++s) {
    const float4 qv = *(const float4*)(q + ((size_t)(s * 32 + b) << 7) + d0);

    float sum = vv.x * tanh_fast(qv.x + hv.x)
              + vv.y * tanh_fast(qv.y + hv.y)
              + vv.z * tanh_fast(qv.z + hv.z)
              + vv.w * tanh_fast(qv.w + hv.w);

    sum += __shfl_xor(sum, 16, 32);
    sum += __shfl_xor(sum,  8, 32);
    sum += __shfl_xor(sum,  4, 32);
    sum += __shfl_xor(sum,  2, 32);
    sum += __shfl_xor(sum,  1, 32);

    if (lane == 0) atts[(size_t)s * 16384 + w] = sum;
  }
}

__global__ void zero_tail(float* __restrict__ p, int n)
{
  int i = blockIdx.x * blockDim.x + threadIdx.x;
  if (i < n) p[i] = 0.0f;
}

// ---------------------------------------------------------------------------
extern "C" void kernel_launch(void* const* d_in, const int* in_sizes, int n_in,
                              void* d_out, int out_size, void* d_ws, size_t ws_size,
                              hipStream_t stream)
{
  const int*   ids  = (const int*)  d_in[0];   // input_ids  [32,64]
  const float* fhid = (const float*)d_in[2];   // first_hidden [32,768]
  const float* ctx  = (const float*)d_in[3];   // context [32,512,768]
  const float* emb  = (const float*)d_in[4];   // emb_table [30522,768]
  const float* Wi   = (const float*)d_in[5];   // [768,128]
  const float* bi   = (const float*)d_in[6];   // [128]
  const float* Wh   = (const float*)d_in[7];   // [768,128]
  const float* bh   = (const float*)d_in[8];   // [128]
  const float* V    = (const float*)d_in[9];   // [128]
  const float* femb = (const float*)d_in[10];  // first_emb [768]
  const float* Wd   = (const float*)d_in[11];  // [1536,768]
  const float* bd   = (const float*)d_in[12];  // [768]

  // ---- workspace carve (all chunks 32B-aligned) ----
  float* hdc   = (float*)d_ws;                        // 16384*128 f32 (8 MB)
  float* h_all = hdc   + (size_t)16384 * 128;         // 65*32*768 f32
  float* q     = h_all + (size_t)65 * 32 * 768;       // 2048*128 f32
  _Float16* WhP = (_Float16*)(q + (size_t)2048 * 128);// 768*128 f16
  _Float16* WiP = WhP + (size_t)768 * 128;            // 768*128 f16
  _Float16* WdP = WiP + (size_t)768 * 128;            // 1536*768 f16

  float* atts = (float*)d_out;                        // 64*32*512
  float* prts = atts + (size_t)64 * 32 * 512;         // 64*32

  // 0) pack weights into WMMA B-fragment f16 layouts
  pack_b_f16<<<(768 * 128 + 255) / 256, 256, 0, stream>>>(Wh, WhP, 768, 128);
  pack_b_f16<<<(768 * 128 + 255) / 256, 256, 0, stream>>>(Wi, WiP, 768, 128);
  pack_b_f16<<<(1536 * 768 + 255) / 256, 256, 0, stream>>>(Wd, WdP, 1536, 768);

  // 1) hdc = context @ Wh + bh : M=16384, N=128 (CT=8 -> 1024 waves)
  wmma_gemm_pk<8><<<128, 256, 0, stream>>>(ctx, 768, WhP, bh, hdc,
                                           16384, 128, 768);

  // 2) full 65-step recurrence, single workgroup, h resident in LDS
  wmma_step_all<<<1, 768, 0, stream>>>(ids, emb, femb, fhid, WdP, bd, h_all);

  // 3) q = h_all[1:65] @ Wi + bi : M=2048, N=128 (CT=8 -> 128 waves)
  wmma_gemm_pk<8><<<16, 256, 0, stream>>>(h_all + (size_t)32 * 768, 768,
                                          WiP, bi, q, 2048, 128, 768);

  // 4) fused attention: 16384 (b,l) waves, 64 steps each
  attn_kernel<<<2048, 256, 0, stream>>>(q, hdc, V, atts);

  // 5) prts = 0
  zero_tail<<<8, 256, 0, stream>>>(prts, 64 * 32);
}
